// SelfAttention2d_72095321030760
// MI455X (gfx1250) — matrix operand order repaired and checked by reading, not compiled, not run on hardware
//
#include <hip/hip_runtime.h>
#include <hip/hip_bf16.h>

// ---------------------------------------------------------------------------
// Self-attention 2D block for MI455X (gfx1250, wave32, WMMA bf16 16x16x32)
//   x:[8,256,64,64] fp32 -> GN(8) -> qkv(1x1) -> flash-attn -> proj(1x1) -> +x
// ---------------------------------------------------------------------------

typedef __attribute__((ext_vector_type(16))) __bf16        v16bf;
typedef __attribute__((ext_vector_type(8)))  float         v8f;
typedef __attribute__((ext_vector_type(4)))  unsigned int  v4u;

union Frag {
  v16bf v;
  v4u   q[2];
};

__device__ __forceinline__ unsigned short f2bf(float f) {
  unsigned u = __float_as_uint(f);
  unsigned r = (u + 0x7FFFu + ((u >> 16) & 1u)) >> 16;  // round-to-nearest-even
  return (unsigned short)r;
}

__device__ __forceinline__ v8f wmma_bf16(const Frag& a, const Frag& b, v8f c) {
  return __builtin_amdgcn_wmma_f32_16x16x32_bf16(
      /*neg_a=*/false, a.v, /*neg_b=*/false, b.v,
      /*c_mod=*/(short)0, c, /*reuse_a=*/false, /*reuse_b=*/false);
}

static constexpr int BB = 8;     // batch
static constexpr int CH = 256;   // channels (= head dim)
static constexpr int NQ = 4096;  // tokens (64*64)

// ---------------------------------------------------------------------------
// 1) GroupNorm statistics: one block per (b, group); 32ch * 4096 = 131072 elems
// ---------------------------------------------------------------------------
__global__ __launch_bounds__(256) void gn_stats_kernel(const float* __restrict__ x,
                                                       float* __restrict__ stats) {
  __shared__ float ssum[256];
  __shared__ float ssq[256];
  const int bg = blockIdx.x;  // 0..63  (groups are channel-contiguous)
  const float* base = x + (size_t)bg * 32 * NQ;
  float s = 0.f, sq = 0.f;
  for (int i = threadIdx.x; i < 32 * NQ; i += 256) {
    float v = base[i];
    s += v;
    sq += v * v;
  }
  ssum[threadIdx.x] = s;
  ssq[threadIdx.x] = sq;
  __syncthreads();
  for (int st = 128; st > 0; st >>= 1) {
    if (threadIdx.x < st) {
      ssum[threadIdx.x] += ssum[threadIdx.x + st];
      ssq[threadIdx.x] += ssq[threadIdx.x + st];
    }
    __syncthreads();
  }
  if (threadIdx.x == 0) {
    const float inv_n = 1.0f / (32.0f * (float)NQ);
    float mean = ssum[0] * inv_n;
    float var = ssq[0] * inv_n - mean * mean;
    stats[bg * 2 + 0] = mean;
    stats[bg * 2 + 1] = rsqrtf(var + 1e-5f);
  }
}

// ---------------------------------------------------------------------------
// 2) GroupNorm apply + transpose + cast: xn[b][n][c] (bf16, c contiguous)
// ---------------------------------------------------------------------------
__global__ __launch_bounds__(256) void gn_apply_kernel(const float* __restrict__ x,
                                                       const float* __restrict__ gamma,
                                                       const float* __restrict__ beta,
                                                       const float* __restrict__ stats,
                                                       unsigned short* __restrict__ xn) {
  const int t = blockIdx.x * 256 + threadIdx.x;   // 8*4096*256 total
  const int c = t & 255;
  const int i = (t >> 8) & 4095;
  const int b = t >> 20;
  const int g = c >> 5;
  const float mean = stats[(b * 8 + g) * 2 + 0];
  const float rstd = stats[(b * 8 + g) * 2 + 1];
  const float v = x[((size_t)b * CH + c) * NQ + i];
  const float y = (v - mean) * rstd * gamma[c] + beta[c];
  xn[((size_t)b * NQ + i) * CH + c] = f2bf(y);
}

// ---------------------------------------------------------------------------
// 3) Weight cast fp32 -> bf16 (w_qkv: 768x256, w_proj: 256x256)
// ---------------------------------------------------------------------------
__global__ __launch_bounds__(256) void cast_w_kernel(const float* __restrict__ wq,
                                                     const float* __restrict__ wp,
                                                     unsigned short* __restrict__ wqb,
                                                     unsigned short* __restrict__ wpb) {
  const int t = blockIdx.x * 256 + threadIdx.x;
  if (t < 768 * 256) {
    wqb[t] = f2bf(wq[t]);
  } else {
    const int u = t - 768 * 256;
    if (u < 256 * 256) wpb[u] = f2bf(wp[u]);
  }
}

// ---------------------------------------------------------------------------
// 4) QKV GEMM: qkv[o,i] = sum_c W[o,c]*xn[c,i]+b ; one wave per 16x16 tile.
//    Routes: o<256 -> Q[b][i][o] ; o<512 -> K[b][i][o-256] ; else V[b][o-512][i]
// ---------------------------------------------------------------------------
__global__ __launch_bounds__(256) void qkv_gemm_kernel(const unsigned short* __restrict__ W,
                                                       const unsigned short* __restrict__ XN,
                                                       const float* __restrict__ bqkv,
                                                       unsigned short* __restrict__ Q,
                                                       unsigned short* __restrict__ K,
                                                       unsigned short* __restrict__ V) {
  const int lane = threadIdx.x & 31;
  const int wv = threadIdx.x >> 5;
  const int task = blockIdx.x * 8 + wv;  // 8*48*256 tasks
  const int nt = task % 256;
  const int mt = (task / 256) % 48;
  const int b = task / (256 * 48);
  const int half = lane >> 4;
  const int l15 = lane & 15;

  v8f acc;
#pragma unroll
  for (int r = 0; r < 8; ++r) acc[r] = 0.f;

  const unsigned short* arow = W + (size_t)(mt * 16 + l15) * CH;
  const unsigned short* bcol = XN + ((size_t)b * NQ + nt * 16 + l15) * CH + 16 * half;
#pragma unroll
  for (int ks = 0; ks < 8; ++ks) {
    Frag a, bf;
    a.q[0] = *(const v4u*)(arow + ks * 32 + 16 * half);
    a.q[1] = *(const v4u*)(arow + ks * 32 + 16 * half + 8);
    bf.q[0] = *(const v4u*)(bcol + ks * 32);
    bf.q[1] = *(const v4u*)(bcol + ks * 32 + 8);
    acc = wmma_bf16(a, bf, acc);
  }

  const int i = nt * 16 + l15;
#pragma unroll
  for (int r = 0; r < 8; ++r) {
    const int o = mt * 16 + r + 8 * half;
    const unsigned short bv = f2bf(acc[r] + bqkv[o]);
    if (o < 256)
      Q[((size_t)b * NQ + i) * CH + o] = bv;
    else if (o < 512)
      K[((size_t)b * NQ + i) * CH + (o - 256)] = bv;
    else
      V[((size_t)b * CH + (o - 512)) * NQ + i] = bv;
  }
}

// ---------------------------------------------------------------------------
// 5) Flash attention: each wave = 16 queries x 256 output channels.
//    Key chunks of 64 (two WMMA K-depths) to amortize online-softmax VALU:
//    the 128-element O rescale + shuffle reductions run once per 64 keys.
// ---------------------------------------------------------------------------
__global__ __launch_bounds__(256) void attn_kernel(const unsigned short* __restrict__ Q,
                                                   const unsigned short* __restrict__ K,
                                                   const unsigned short* __restrict__ V,
                                                   unsigned short* __restrict__ ATT) {
  __shared__ unsigned short pbuf[8][16 * 64];  // per-wave P tile (bf16), 2 KB
  const int lane = threadIdx.x & 31;
  const int wv = threadIdx.x >> 5;
  const int task = blockIdx.x * 8 + wv;  // 8 batches * 256 qtiles
  const int b = task >> 8;
  const int qt = task & 255;
  const int half = lane >> 4;
  const int l15 = lane & 15;
  const float scale = 0.0625f;  // 1/sqrt(256)

  // Preload Q A-fragments for all 8 K-steps (rows = queries, K = channels)
  Frag aq[8];
  {
    const unsigned short* rp = Q + ((size_t)b * NQ + qt * 16 + l15) * CH;
#pragma unroll
    for (int ks = 0; ks < 8; ++ks) {
      aq[ks].q[0] = *(const v4u*)(rp + ks * 32 + 16 * half);
      aq[ks].q[1] = *(const v4u*)(rp + ks * 32 + 16 * half + 8);
    }
  }

  v8f o[16];
#pragma unroll
  for (int ct = 0; ct < 16; ++ct)
#pragma unroll
    for (int r = 0; r < 8; ++r) o[ct][r] = 0.f;

  float m[8], l[8];
#pragma unroll
  for (int r = 0; r < 8; ++r) {
    m[r] = -1e30f;
    l[r] = 0.f;
  }

  for (int j0 = 0; j0 < NQ; j0 += 64) {
    // ---- S = (Q^T K) * scale : four 16x16 tiles over this 64-key chunk
    v8f s[4];
#pragma unroll
    for (int jt = 0; jt < 4; ++jt) {
#pragma unroll
      for (int r = 0; r < 8; ++r) s[jt][r] = 0.f;
      const unsigned short* kp =
          K + ((size_t)b * NQ + j0 + jt * 16 + l15) * CH + 16 * half;
#pragma unroll
      for (int ks = 0; ks < 8; ++ks) {
        Frag bk;
        bk.q[0] = *(const v4u*)(kp + ks * 32);
        bk.q[1] = *(const v4u*)(kp + ks * 32 + 8);
        s[jt] = wmma_bf16(aq[ks], bk, s[jt]);
      }
    }

    // ---- online softmax update (row = r + 8*half, cols across 16 lanes)
#pragma unroll
    for (int r = 0; r < 8; ++r) {
      float p0 = s[0][r] * scale;
      float p1 = s[1][r] * scale;
      float p2 = s[2][r] * scale;
      float p3 = s[3][r] * scale;
      float rm = fmaxf(fmaxf(p0, p1), fmaxf(p2, p3));
#pragma unroll
      for (int off = 8; off > 0; off >>= 1) rm = fmaxf(rm, __shfl_xor(rm, off, 32));
      const float mn = fmaxf(m[r], rm);
      const float corr = __expf(m[r] - mn);
      p0 = __expf(p0 - mn);
      p1 = __expf(p1 - mn);
      p2 = __expf(p2 - mn);
      p3 = __expf(p3 - mn);
      float rs = (p0 + p1) + (p2 + p3);
#pragma unroll
      for (int off = 8; off > 0; off >>= 1) rs += __shfl_xor(rs, off, 32);
      l[r] = l[r] * corr + rs;
      m[r] = mn;
#pragma unroll
      for (int ct = 0; ct < 16; ++ct) o[ct][r] *= corr;
      s[0][r] = p0;
      s[1][r] = p1;
      s[2][r] = p2;
      s[3][r] = p3;
    }

    // ---- transpose P (16x64) from C-layout to A-layout through LDS (bf16)
    unsigned short* pb = pbuf[wv];
#pragma unroll
    for (int r = 0; r < 8; ++r) {
      const int row = r + 8 * half;
#pragma unroll
      for (int jt = 0; jt < 4; ++jt)
        pb[row * 64 + jt * 16 + l15] = f2bf(s[jt][r]);
    }
    __builtin_amdgcn_wave_barrier();
    Frag pa[2];  // K-chunks 0..31 and 32..63 of the P tile
    {
      const unsigned short* rp = pb + l15 * 64;
#pragma unroll
      for (int kc = 0; kc < 2; ++kc) {
        pa[kc].q[0] = *(const v4u*)(rp + kc * 32 + 16 * half);
        pa[kc].q[1] = *(const v4u*)(rp + kc * 32 + 16 * half + 8);
      }
    }
    __builtin_amdgcn_wave_barrier();

    // ---- O += P x V^T  (B operand: V[b][c][j], j contiguous)
#pragma unroll
    for (int ct = 0; ct < 16; ++ct) {
      const unsigned short* vp =
          V + ((size_t)b * CH + ct * 16 + l15) * NQ + j0 + 16 * half;
#pragma unroll
      for (int kc = 0; kc < 2; ++kc) {
        Frag bv;
        bv.q[0] = *(const v4u*)(vp + kc * 32);
        bv.q[1] = *(const v4u*)(vp + kc * 32 + 8);
        o[ct] = wmma_bf16(pa[kc], bv, o[ct]);
      }
    }
  }

  // ---- finalize: divide by row sum, write ATT[b][i][c] (bf16)
#pragma unroll
  for (int r = 0; r < 8; ++r) {
    const float inv = 1.0f / l[r];
    const int row = qt * 16 + r + 8 * half;
#pragma unroll
    for (int ct = 0; ct < 16; ++ct) {
      ATT[((size_t)b * NQ + row) * CH + ct * 16 + l15] = f2bf(o[ct][r] * inv);
    }
  }
}

// ---------------------------------------------------------------------------
// 6) Proj GEMM + bias + residual: out[b][o][i] = x + b_proj[o] + Wp @ att
// ---------------------------------------------------------------------------
__global__ __launch_bounds__(256) void proj_gemm_kernel(const unsigned short* __restrict__ Wp,
                                                        const unsigned short* __restrict__ ATT,
                                                        const float* __restrict__ bproj,
                                                        const float* __restrict__ x,
                                                        float* __restrict__ out) {
  const int lane = threadIdx.x & 31;
  const int wv = threadIdx.x >> 5;
  const int task = blockIdx.x * 8 + wv;  // 8*16*256 tasks
  const int nt = task % 256;
  const int mt = (task / 256) % 16;
  const int b = task / (256 * 16);
  const int half = lane >> 4;
  const int l15 = lane & 15;

  v8f acc;
#pragma unroll
  for (int r = 0; r < 8; ++r) acc[r] = 0.f;

  const unsigned short* arow = Wp + (size_t)(mt * 16 + l15) * CH;
  const unsigned short* bcol = ATT + ((size_t)b * NQ + nt * 16 + l15) * CH + 16 * half;
#pragma unroll
  for (int ks = 0; ks < 8; ++ks) {
    Frag a, bf;
    a.q[0] = *(const v4u*)(arow + ks * 32 + 16 * half);
    a.q[1] = *(const v4u*)(arow + ks * 32 + 16 * half + 8);
    bf.q[0] = *(const v4u*)(bcol + ks * 32);
    bf.q[1] = *(const v4u*)(bcol + ks * 32 + 8);
    acc = wmma_bf16(a, bf, acc);
  }

  const int i = nt * 16 + l15;
#pragma unroll
  for (int r = 0; r < 8; ++r) {
    const int oc = mt * 16 + r + 8 * half;
    const size_t idx = ((size_t)b * CH + oc) * NQ + i;
    out[idx] = x[idx] + bproj[oc] + acc[r];
  }
}

// ---------------------------------------------------------------------------
// Host launcher
// ---------------------------------------------------------------------------
extern "C" void kernel_launch(void* const* d_in, const int* in_sizes, int n_in,
                              void* d_out, int out_size, void* d_ws, size_t ws_size,
                              hipStream_t stream) {
  const float* x = (const float*)d_in[0];
  const float* gamma = (const float*)d_in[1];
  const float* beta = (const float*)d_in[2];
  const float* w_qkv = (const float*)d_in[3];
  const float* b_qkv = (const float*)d_in[4];
  const float* w_proj = (const float*)d_in[5];
  const float* b_proj = (const float*)d_in[6];
  float* out = (float*)d_out;

  char* ws = (char*)d_ws;
  const size_t BIG = (size_t)BB * NQ * CH * 2;  // 16.78 MB each (bf16)
  float* stats = (float*)ws;                                    // 64*2 f32
  unsigned short* wqb = (unsigned short*)(ws + 4096);           // 768*256 bf16
  unsigned short* wpb = (unsigned short*)(ws + 4096 + 768 * 256 * 2);
  size_t off = 4096 + 768 * 256 * 2 + 256 * 256 * 2;
  unsigned short* xn = (unsigned short*)(ws + off); off += BIG; // [b][n][c]
  unsigned short* Q  = (unsigned short*)(ws + off); off += BIG; // [b][n][c]
  unsigned short* K  = (unsigned short*)(ws + off); off += BIG; // [b][n][c]
  unsigned short* V  = (unsigned short*)(ws + off); off += BIG; // [b][c][n]
  unsigned short* AT = (unsigned short*)(ws + off); off += BIG; // [b][n][c]

  gn_stats_kernel<<<BB * 8, 256, 0, stream>>>(x, stats);
  gn_apply_kernel<<<(BB * NQ * CH) / 256, 256, 0, stream>>>(x, gamma, beta, stats, xn);
  cast_w_kernel<<<(768 * 256 + 256 * 256) / 256, 256, 0, stream>>>(w_qkv, w_proj, wqb, wpb);
  qkv_gemm_kernel<<<(BB * 48 * 256) / 8, 256, 0, stream>>>(wqb, xn, b_qkv, Q, K, V);
  attn_kernel<<<(BB * 256) / 8, 256, 0, stream>>>(Q, K, V, AT);
  proj_gemm_kernel<<<(BB * 16 * 256) / 8, 256, 0, stream>>>(wpb, AT, b_proj, x, out);
}
